// CrossAtt_27333171872246
// MI455X (gfx1250) — matrix operand order repaired and checked by reading, not compiled, not run on hardware
//
#include <hip/hip_runtime.h>
#include <math.h>

typedef __attribute__((ext_vector_type(16))) _Float16 v16h;
typedef __attribute__((ext_vector_type(8)))  _Float16 v8h;
typedef __attribute__((ext_vector_type(8)))  float    v8f;

#define NPIX 4096   // 64*64
#define EPSV 1e-5f

// ---------------------------------------------------------------------------
// WMMA helpers. Layouts per CDNA5 ISA (wave32):
//  A (16x32 f16): row = lane&15, elem i -> k = (i&7) + ((i>>3)<<4) + (lane>>4)*8
//  B (32x16 f16): col = lane&15, elem i -> k = (i&7) + ((i>>3)<<4) + (lane>>4)*8
//  C (16x16 f32): col = lane&15, reg r  -> row = r + (lane>>4)*8
// ---------------------------------------------------------------------------
static __device__ __forceinline__ v8f wmma_f32(v16h a, v16h b, v8f c) {
  return __builtin_amdgcn_wmma_f32_16x16x32_f16(false, a, false, b, (short)0, c,
                                                false, false);
}

// A operand: base points at row start (+ k-chunk offset); two contiguous 8-half runs
static __device__ __forceinline__ v16h load_a16(const _Float16* base, int half) {
  const v8h lo = *(const v8h*)(base + half * 8);
  const v8h hi = *(const v8h*)(base + 16 + half * 8);
  v16h a;
#pragma unroll
  for (int i = 0; i < 8; ++i) { a[i] = lo[i]; a[i + 8] = hi[i]; }
  return a;
}

// B operand: base points at this lane's column; two contiguous 8-half runs
static __device__ __forceinline__ v16h load_b16(const _Float16* base, int half) {
  const v8h lo = *(const v8h*)(base + half * 8);
  const v8h hi = *(const v8h*)(base + 16 + half * 8);
  v16h b;
#pragma unroll
  for (int i = 0; i < 8; ++i) { b[i] = lo[i]; b[i + 8] = hi[i]; }
  return b;
}

// ---------------------------------------------------------------------------
// Conversions
// ---------------------------------------------------------------------------
// input [B][256][4096] fp32 -> [B][4096][256] f16 (transposed for GEMM-B)
__global__ void cvt_transpose_in(const float* __restrict__ src,
                                 _Float16* __restrict__ dstT) {
  int idx = blockIdx.x * blockDim.x + threadIdx.x;      // n fastest
  int n = idx & 4095;
  int c = (idx >> 12) & 255;
  int b = idx >> 20;
  dstT[((size_t)b * NPIX + n) * 256 + c] = (_Float16)src[idx];
}

__global__ void cvt_f32_f16(const float* __restrict__ s, _Float16* __restrict__ d,
                            int n) {
  int i = blockIdx.x * blockDim.x + threadIdx.x;
  if (i < n) d[i] = (_Float16)s[i];
}

// cat w1 [128][512][3][3] -> [9][128][512] f16 (tap-major for 9-tap GEMM)
__global__ void reorder_wcat(const float* __restrict__ w, _Float16* __restrict__ wh) {
  int i = blockIdx.x * blockDim.x + threadIdx.x;
  if (i >= 128 * 512 * 9) return;
  int tap = i % 9;
  int ic  = (i / 9) % 512;
  int oc  = i / (9 * 512);
  wh[((size_t)tap * 128 + oc) * 512 + ic] = (_Float16)w[i];
}

// ---------------------------------------------------------------------------
// Ghost primary 1x1 conv as GEMM:  out[M][N] = W[M][256] * X[256][N], + BN + ReLU
// X given transposed as xT [B][N][256].  Writes x1 buffer [B][M][NPIX] (dw input)
// and scatters into attention operand buffers.
//   mode 0: planar dst [B][256][NPIX] at channel offset coff   (V matrices)
//   mode 1: transposed dst [B][NPIX][ld] at column offset coff (QT/KT)
// ---------------------------------------------------------------------------
__global__ __launch_bounds__(128) void pw_gemm_kernel(
    const _Float16* __restrict__ w_h, const float* __restrict__ g,
    const float* __restrict__ bb, const _Float16* __restrict__ xT,
    _Float16* __restrict__ x1, _Float16* __restrict__ dst, int M, int mode,
    int coff, int ld) {
  const int lane = threadIdx.x & 31, wv = threadIdx.x >> 5;
  const int b = blockIdx.y;
  const int tile = blockIdx.x * 4 + wv;
  const int Mt = M >> 4;
  const int nt = tile & 255, mt = tile >> 8;
  if (mt >= Mt) return;
  const int n0 = nt * 16, oc0 = mt * 16;
  const int half = lane >> 4, col = lane & 15;

  const _Float16* arow = w_h + (size_t)(oc0 + col) * 256;
  const _Float16* brow = xT + ((size_t)b * NPIX + n0 + col) * 256;
  v8f acc = {};
#pragma unroll
  for (int kk = 0; kk < 256; kk += 32)
    acc = wmma_f32(load_a16(arow + kk, half), load_b16(brow + kk, half), acc);

  _Float16* x1b = x1 + (size_t)b * M * NPIX;
  const int n = n0 + col;
#pragma unroll
  for (int r = 0; r < 8; ++r) {
    int oc = oc0 + r + (half << 3);
    float sc = g[oc] * rsqrtf(1.0f + EPSV);
    float v  = acc[r] * sc + bb[oc];
    v = v > 0.f ? v : 0.f;
    _Float16 h = (_Float16)v;
    x1b[(size_t)oc * NPIX + n] = h;
    if (mode == 0)
      dst[((size_t)b * 256 + coff + oc) * NPIX + n] = h;
    else
      dst[((size_t)b * NPIX + n) * ld + coff + oc] = h;
  }
}

// ---------------------------------------------------------------------------
// Ghost depthwise 3x3 + BN + ReLU (VALU; tiny fraction of FLOPs)
// ---------------------------------------------------------------------------
__global__ void dw3x3_kernel(const _Float16* __restrict__ x1,
                             const float* __restrict__ w2,
                             const float* __restrict__ g,
                             const float* __restrict__ bb,
                             _Float16* __restrict__ dst, int M, int mode,
                             int coff, int ld) {
  int idx = blockIdx.x * blockDim.x + threadIdx.x;
  int b = blockIdx.y;
  if (idx >= M * NPIX) return;
  int n = idx & 4095, c = idx >> 12;
  int y = n >> 6, x = n & 63;
  const _Float16* xc = x1 + ((size_t)b * M + c) * NPIX;
  float sum = 0.f;
#pragma unroll
  for (int dy = -1; dy <= 1; ++dy)
#pragma unroll
    for (int dx = -1; dx <= 1; ++dx) {
      int yy = y + dy, xx = x + dx;
      if (yy >= 0 && yy < 64 && xx >= 0 && xx < 64)
        sum += w2[c * 9 + (dy + 1) * 3 + (dx + 1)] * (float)xc[yy * 64 + xx];
    }
  float sc = g[c] * rsqrtf(1.0f + EPSV);
  float v = sum * sc + bb[c];
  v = v > 0.f ? v : 0.f;
  _Float16 h = (_Float16)v;
  if (mode == 0)
    dst[((size_t)b * 256 + coff + c) * NPIX + n] = h;
  else
    dst[((size_t)b * NPIX + n) * ld + coff + c] = h;
}

// ---------------------------------------------------------------------------
// Flash attention, one wave per 16-query tile.
//   S = Q(16x64) * K^T   via WMMA, online softmax (shfl_xor within 16-lane
//   halves), P transposed C-layout -> A-layout through LDS, O += P * V^T.
// Writes out = gamma*O + input (fp32, [B][256][NPIX]) and catT f16 [B][NPIX][512].
// ---------------------------------------------------------------------------
__global__ __launch_bounds__(128) void attn_kernel(
    const _Float16* __restrict__ qT, const _Float16* __restrict__ kT,
    const _Float16* __restrict__ vP, const float* __restrict__ inp,
    const float* __restrict__ gamma_p, float* __restrict__ outP,
    _Float16* __restrict__ catT, int coff) {
  const int lane = threadIdx.x & 31, wv = threadIdx.x >> 5;
  const int b = blockIdx.y;
  const int q0 = (blockIdx.x * 4 + wv) * 16;
  const int half = lane >> 4, col = lane & 15;
  const _Float16* qb = qT + (size_t)b * NPIX * 64;
  const _Float16* kb = kT + (size_t)b * NPIX * 64;
  const _Float16* vb = vP + (size_t)b * 256 * NPIX;

  __shared__ __align__(16) _Float16 sP[4][16 * 32];
  _Float16* sp = sP[wv];

  const v16h aq0 = load_a16(qb + (size_t)(q0 + col) * 64, half);
  const v16h aq1 = load_a16(qb + (size_t)(q0 + col) * 64 + 32, half);

  v8f acc[16];
#pragma unroll
  for (int t = 0; t < 16; ++t) acc[t] = (v8f){};
  float mrow[8], lrow[8];
#pragma unroll
  for (int r = 0; r < 8; ++r) { mrow[r] = -1e30f; lrow[r] = 0.f; }

  for (int m0 = 0; m0 < NPIX; m0 += 32) {
    const _Float16* kr0 = kb + (size_t)(m0 + col) * 64;
    const _Float16* kr1 = kb + (size_t)(m0 + 16 + col) * 64;
    v8f s0 = {}, s1 = {};
    s0 = wmma_f32(aq0, load_b16(kr0, half), s0);
    s0 = wmma_f32(aq1, load_b16(kr0 + 32, half), s0);
    s1 = wmma_f32(aq0, load_b16(kr1, half), s1);
    s1 = wmma_f32(aq1, load_b16(kr1 + 32, half), s1);

#pragma unroll
    for (int r = 0; r < 8; ++r) {
      float mx = fmaxf(s0[r], s1[r]);
      mx = fmaxf(mx, __shfl_xor(mx, 1, 16));
      mx = fmaxf(mx, __shfl_xor(mx, 2, 16));
      mx = fmaxf(mx, __shfl_xor(mx, 4, 16));
      mx = fmaxf(mx, __shfl_xor(mx, 8, 16));
      float mn = fmaxf(mrow[r], mx);
      float alpha = __expf(mrow[r] - mn);
      float p0 = __expf(s0[r] - mn);
      float p1 = __expf(s1[r] - mn);
      float ls = p0 + p1;
      ls += __shfl_xor(ls, 1, 16);
      ls += __shfl_xor(ls, 2, 16);
      ls += __shfl_xor(ls, 4, 16);
      ls += __shfl_xor(ls, 8, 16);
      lrow[r] = lrow[r] * alpha + ls;
      mrow[r] = mn;
      sp[(r + half * 8) * 32 + col] = (_Float16)p0;
      sp[(r + half * 8) * 32 + 16 + col] = (_Float16)p1;
#pragma unroll
      for (int t = 0; t < 16; ++t) acc[t][r] *= alpha;
    }
    // cross-lane RAW through LDS within the wave: wait for ds stores
    asm volatile("s_wait_dscnt 0x0" ::: "memory");
    const v16h ap = load_a16(sp + col * 32, half);
#pragma unroll
    for (int t = 0; t < 16; ++t) {
      const _Float16* vr = vb + (size_t)(t * 16 + col) * NPIX + m0;
      acc[t] = wmma_f32(ap, load_b16(vr, half), acc[t]);
    }
  }

  const float gamma = *gamma_p;
  const float* ib = inp + (size_t)b * 256 * NPIX;
  float* ob = outP + (size_t)b * 256 * NPIX;
  _Float16* cb = catT + (size_t)b * NPIX * 512;
#pragma unroll
  for (int r = 0; r < 8; ++r) {
    int q = q0 + r + half * 8;
    float rl = 1.0f / lrow[r];
#pragma unroll
    for (int t = 0; t < 16; ++t) {
      int ch = t * 16 + col;
      float o = gamma * (acc[t][r] * rl) + ib[(size_t)ch * NPIX + q];
      ob[(size_t)ch * NPIX + q] = o;
      cb[(size_t)q * 512 + coff + ch] = (_Float16)o;
    }
  }
}

// ---------------------------------------------------------------------------
// cat Ghost primary 3x3 conv over 512 channels: 9-tap implicit GEMM.
// Fuses inner BN+ReLU (ghost) and outer BN+ReLU (feat_sum, channels 0..127).
// ---------------------------------------------------------------------------
__global__ __launch_bounds__(128) void cat_conv_kernel(
    const _Float16* __restrict__ wh,    // [9][128][512]
    const _Float16* __restrict__ catT,  // [B][NPIX][512]
    const float* __restrict__ g1, const float* __restrict__ b1,
    const float* __restrict__ go, const float* __restrict__ bo,
    _Float16* __restrict__ x1cat,       // [B][128][NPIX]
    float* __restrict__ feat) {         // [B][256][NPIX]
  const int lane = threadIdx.x & 31, wv = threadIdx.x >> 5;
  const int b = blockIdx.y;
  const int tile = blockIdx.x * 4 + wv;       // 8*256 tiles per batch
  const int nt = tile & 255, mt = tile >> 8;
  const int n0 = nt * 16, oc0 = mt * 16;
  const int half = lane >> 4, col = lane & 15;
  const int n = n0 + col, y = n >> 6, x = n & 63;
  const _Float16* cbT = catT + (size_t)b * NPIX * 512;

  v8f acc = {};
#pragma unroll
  for (int dy = -1; dy <= 1; ++dy) {
    int yy = y + dy;
#pragma unroll
    for (int dx = -1; dx <= 1; ++dx) {
      int xx = x + dx;
      bool ok = (yy >= 0) & (yy < 64) & (xx >= 0) & (xx < 64);
      int pix = ok ? (yy * 64 + xx) : 0;
      const int tap = (dy + 1) * 3 + (dx + 1);
      const _Float16* arow = wh + ((size_t)tap * 128 + oc0 + col) * 512;
      const _Float16* brow = cbT + (size_t)pix * 512;
#pragma unroll
      for (int kk = 0; kk < 512; kk += 32) {
        v16h a = load_a16(arow + kk, half);
        v16h bm = {};
        if (ok) bm = load_b16(brow + kk, half);
        acc = wmma_f32(a, bm, acc);
      }
    }
  }
  _Float16* x1b = x1cat + (size_t)b * 128 * NPIX;
  float* fb = feat + (size_t)b * 256 * NPIX;
#pragma unroll
  for (int r = 0; r < 8; ++r) {
    int oc = oc0 + r + (half << 3);
    float s1 = g1[oc] * rsqrtf(1.0f + EPSV);
    float v  = acc[r] * s1 + b1[oc];
    v = v > 0.f ? v : 0.f;                        // ghost inner relu
    x1b[(size_t)oc * NPIX + n] = (_Float16)v;
    float so = go[oc] * rsqrtf(1.0f + EPSV);
    float f  = v * so + bo[oc];
    f = f > 0.f ? f : 0.f;                        // outer relu
    fb[(size_t)oc * NPIX + n] = f;
  }
}

// cat Ghost depthwise 3x3 + inner BN/ReLU + outer BN/ReLU -> feat channels 128..255
__global__ void cat_dw_kernel(const _Float16* __restrict__ x1,
                              const float* __restrict__ w2,
                              const float* __restrict__ g2,
                              const float* __restrict__ b2,
                              const float* __restrict__ go,
                              const float* __restrict__ bo,
                              float* __restrict__ feat) {
  int idx = blockIdx.x * blockDim.x + threadIdx.x;
  int b = blockIdx.y;
  if (idx >= 128 * NPIX) return;
  int n = idx & 4095, c = idx >> 12;
  int y = n >> 6, x = n & 63;
  const _Float16* xc = x1 + ((size_t)b * 128 + c) * NPIX;
  float sum = 0.f;
#pragma unroll
  for (int dy = -1; dy <= 1; ++dy)
#pragma unroll
    for (int dx = -1; dx <= 1; ++dx) {
      int yy = y + dy, xx = x + dx;
      if (yy >= 0 && yy < 64 && xx >= 0 && xx < 64)
        sum += w2[c * 9 + (dy + 1) * 3 + (dx + 1)] * (float)xc[yy * 64 + xx];
    }
  float s2 = g2[c] * rsqrtf(1.0f + EPSV);
  float v = sum * s2 + b2[c];
  v = v > 0.f ? v : 0.f;
  int oc = 128 + c;
  float so = go[oc] * rsqrtf(1.0f + EPSV);
  float f = v * so + bo[oc];
  f = f > 0.f ? f : 0.f;
  feat[((size_t)b * 256 + oc) * NPIX + n] = f;
}

// ---------------------------------------------------------------------------
extern "C" void kernel_launch(void* const* d_in, const int* in_sizes, int n_in,
                              void* d_out, int out_size, void* d_ws,
                              size_t ws_size, hipStream_t stream) {
  const int B = 2;
  const float* input1 = (const float*)d_in[0];
  const float* input2 = (const float*)d_in[1];
  // ghost param index bases: w1,g1,b1,w2,g2,b2
  const int Q1 = 2, K1 = 8, V1 = 14, Q2 = 20, K2 = 26, V2 = 32, CT = 38;
  const float* cat_g = (const float*)d_in[44];
  const float* cat_b = (const float*)d_in[45];
  const float* gamma = (const float*)d_in[46];

  float* feat = (float*)d_out;                    // [2][256][4096]
  float* out1 = feat + (size_t)B * 256 * NPIX;    // [2][256][4096]
  float* out2 = out1 + (size_t)B * 256 * NPIX;

  char* ws = (char*)d_ws;
  size_t off = 0;
  auto carve = [&](size_t bytes) {
    void* p = ws + off;
    off += (bytes + 255) & ~(size_t)255;
    return p;
  };
  _Float16* in1T  = (_Float16*)carve((size_t)B * NPIX * 256 * 2);
  _Float16* in2T  = (_Float16*)carve((size_t)B * NPIX * 256 * 2);
  _Float16* QT    = (_Float16*)carve((size_t)B * NPIX * 64 * 2);
  _Float16* KT1   = (_Float16*)carve((size_t)B * NPIX * 64 * 2);
  _Float16* KT2   = (_Float16*)carve((size_t)B * NPIX * 64 * 2);
  _Float16* Vm1   = (_Float16*)carve((size_t)B * 256 * NPIX * 2);
  _Float16* Vm2   = (_Float16*)carve((size_t)B * 256 * NPIX * 2);
  _Float16* x1q1  = (_Float16*)carve((size_t)B * 16 * NPIX * 2);
  _Float16* x1k1  = (_Float16*)carve((size_t)B * 32 * NPIX * 2);
  _Float16* x1v1  = (_Float16*)carve((size_t)B * 128 * NPIX * 2);
  _Float16* x1q2  = (_Float16*)carve((size_t)B * 16 * NPIX * 2);
  _Float16* x1k2  = (_Float16*)carve((size_t)B * 32 * NPIX * 2);
  _Float16* x1v2  = (_Float16*)carve((size_t)B * 128 * NPIX * 2);
  _Float16* catT  = (_Float16*)carve((size_t)B * NPIX * 512 * 2);
  _Float16* x1cat = (_Float16*)carve((size_t)B * 128 * NPIX * 2);
  _Float16* whq1  = (_Float16*)carve(16 * 256 * 2);
  _Float16* whk1  = (_Float16*)carve(32 * 256 * 2);
  _Float16* whv1  = (_Float16*)carve(128 * 256 * 2);
  _Float16* whq2  = (_Float16*)carve(16 * 256 * 2);
  _Float16* whk2  = (_Float16*)carve(32 * 256 * 2);
  _Float16* whv2  = (_Float16*)carve(128 * 256 * 2);
  _Float16* whcat = (_Float16*)carve((size_t)9 * 128 * 512 * 2);

  // --- conversions ---
  cvt_transpose_in<<<(B * 256 * NPIX) / 256, 256, 0, stream>>>(input1, in1T);
  cvt_transpose_in<<<(B * 256 * NPIX) / 256, 256, 0, stream>>>(input2, in2T);
  auto cvtw = [&](int base, _Float16* dst, int m) {
    int n = m * 256;
    cvt_f32_f16<<<(n + 255) / 256, 256, 0, stream>>>((const float*)d_in[base], dst, n);
  };
  cvtw(Q1, whq1, 16); cvtw(K1, whk1, 32); cvtw(V1, whv1, 128);
  cvtw(Q2, whq2, 16); cvtw(K2, whk2, 32); cvtw(V2, whv2, 128);
  reorder_wcat<<<(128 * 512 * 9 + 255) / 256, 256, 0, stream>>>(
      (const float*)d_in[CT], whcat);

  // --- ghost primary 1x1 convs (WMMA GEMM) ---
  auto pw = [&](const _Float16* wh, int base, const _Float16* xT, _Float16* x1,
                _Float16* dst, int M, int mode, int coff, int ld) {
    dim3 g((M >> 4) * 256 / 4, B);
    pw_gemm_kernel<<<g, 128, 0, stream>>>(wh, (const float*)d_in[base + 1],
                                          (const float*)d_in[base + 2], xT, x1,
                                          dst, M, mode, coff, ld);
  };
  pw(whq1, Q1, in1T, x1q1, QT, 16, 1, 0, 64);
  pw(whk1, K1, in1T, x1k1, KT1, 32, 1, 0, 64);
  pw(whv1, V1, in1T, x1v1, Vm1, 128, 0, 0, 0);
  pw(whq2, Q2, in2T, x1q2, QT, 16, 1, 32, 64);
  pw(whk2, K2, in2T, x1k2, KT2, 32, 1, 0, 64);
  pw(whv2, V2, in2T, x1v2, Vm2, 128, 0, 0, 0);

  // --- ghost depthwise 3x3 ---
  auto dw = [&](const _Float16* x1, int base, _Float16* dst, int M, int mode,
                int coff, int ld) {
    dim3 g((M * NPIX + 255) / 256, B);
    dw3x3_kernel<<<g, 256, 0, stream>>>(x1, (const float*)d_in[base + 3],
                                        (const float*)d_in[base + 4],
                                        (const float*)d_in[base + 5], dst, M,
                                        mode, coff, ld);
  };
  dw(x1q1, Q1, QT, 16, 1, 16, 64);
  dw(x1k1, K1, KT1, 32, 1, 32, 64);
  dw(x1v1, V1, Vm1, 128, 0, 128, 0);
  dw(x1q2, Q2, QT, 16, 1, 48, 64);
  dw(x1k2, K2, KT2, 32, 1, 32, 64);
  dw(x1v2, V2, Vm2, 128, 0, 128, 0);

  // --- flash attention (both branches) ---
  {
    dim3 g(256 / 4, B);
    attn_kernel<<<g, 128, 0, stream>>>(QT, KT1, Vm1, input1, gamma, out1, catT, 0);
    attn_kernel<<<g, 128, 0, stream>>>(QT, KT2, Vm2, input2, gamma, out2, catT, 256);
  }

  // --- cat ghost: 3x3 conv (9-tap WMMA GEMM) + depthwise ---
  {
    dim3 g(8 * 256 / 4, B);
    cat_conv_kernel<<<g, 128, 0, stream>>>(whcat, catT, (const float*)d_in[CT + 1],
                                           (const float*)d_in[CT + 2], cat_g,
                                           cat_b, x1cat, feat);
  }
  {
    dim3 g((128 * NPIX + 255) / 256, B);
    cat_dw_kernel<<<g, 256, 0, stream>>>(x1cat, (const float*)d_in[CT + 3],
                                         (const float*)d_in[CT + 4],
                                         (const float*)d_in[CT + 5], cat_g,
                                         cat_b, feat);
  }
}
